// MultiScalePointNet2Encoder_78365973283331
// MI455X (gfx1250) — matrix operand order repaired and checked by reading, not compiled, not run on hardware
//
#include <hip/hip_runtime.h>
#include <hip/hip_bf16.h>

typedef __attribute__((ext_vector_type(16))) _Float16 v16h;
typedef __attribute__((ext_vector_type(8)))  float    v8f;
typedef int v4i32 __attribute__((__vector_size__(16)));

#define BB 4
#define NN 4096

#if __has_builtin(__builtin_amdgcn_global_load_async_to_lds_b128) && \
    __has_builtin(__builtin_amdgcn_s_wait_asynccnt)
#define USE_ASYNC_LDS 1
#endif

static __host__ __device__ inline int alignup(int x, int a) { return (x + a - 1) / a * a; }

// -------------------- transpose pc (B,3,N) -> xyz (B,N,3) --------------------
__global__ void k_split_pc(const float* __restrict__ pc, float* __restrict__ xyz) {
  int t = blockIdx.x * blockDim.x + threadIdx.x;
  if (t >= BB * NN) return;
  int b = t / NN, n = t % NN;
#pragma unroll
  for (int c = 0; c < 3; c++) xyz[(size_t)(b * NN + n) * 3 + c] = pc[(size_t)(b * 3 + c) * NN + n];
}

// -------------------- farthest point sampling: 1 block per batch --------------------
__global__ __launch_bounds__(256) void k_fps(const float* __restrict__ xyz, int Nin, int npoint,
                                             int* __restrict__ out) {
  __shared__ float dist[NN];
  __shared__ float rv[256];
  __shared__ int   ri[256];
  __shared__ int   far_s;
  const int b = blockIdx.x, tid = threadIdx.x;
  const float* xp = xyz + (size_t)b * Nin * 3;
  for (int j = tid; j < Nin; j += 256) dist[j] = 1e10f;
  if (tid == 0) far_s = 0;
  __syncthreads();
  for (int it = 0; it < npoint; it++) {
    const int far = far_s;
    if (tid == 0) out[b * npoint + it] = far;
    const float cx = xp[far * 3 + 0], cy = xp[far * 3 + 1], cz = xp[far * 3 + 2];
    float best = -1.0f; int bidx = 0;
    for (int j = tid; j < Nin; j += 256) {
      float dx = xp[j * 3 + 0] - cx, dy = xp[j * 3 + 1] - cy, dz = xp[j * 3 + 2] - cz;
      float d = dx * dx + dy * dy + dz * dz;
      float nd = fminf(dist[j], d);
      dist[j] = nd;
      if (nd > best) { best = nd; bidx = j; }
    }
    rv[tid] = best; ri[tid] = bidx;
    __syncthreads();
    for (int st = 128; st > 0; st >>= 1) {
      if (tid < st) {
        if (rv[tid + st] > rv[tid] || (rv[tid + st] == rv[tid] && ri[tid + st] < ri[tid])) {
          rv[tid] = rv[tid + st]; ri[tid] = ri[tid + st];
        }
      }
      __syncthreads();
    }
    if (tid == 0) far_s = ri[0];
    __syncthreads();
  }
}

// -------------------- gather sampled centers --------------------
__global__ void k_gather_nx(const float* __restrict__ xyz, const int* __restrict__ idx,
                            int Nin, int S, float* __restrict__ nx) {
  int t = blockIdx.x * blockDim.x + threadIdx.x;
  if (t >= BB * S) return;
  int b = t / S;
  int id = idx[t];
#pragma unroll
  for (int c = 0; c < 3; c++) nx[(size_t)t * 3 + c] = xyz[((size_t)b * Nin + id) * 3 + c];
}

// -------------------- ball query: first k in-range indices, pad with first --------------------
__global__ void k_ballq(const float* __restrict__ xyz, const float* __restrict__ nx,
                        int Nin, int S, float r2, int k, int* __restrict__ nbr) {
  int t = blockIdx.x * blockDim.x + threadIdx.x;
  if (t >= BB * S) return;
  int b = t / S;
  float cx = nx[(size_t)t * 3 + 0], cy = nx[(size_t)t * 3 + 1], cz = nx[(size_t)t * 3 + 2];
  const float* xp = xyz + (size_t)b * Nin * 3;
  int cnt = 0, first = 0; bool got = false;
  for (int j = 0; j < Nin && cnt < k; j++) {
    float dx = xp[j * 3 + 0] - cx, dy = xp[j * 3 + 1] - cy, dz = xp[j * 3 + 2] - cz;
    float d = dx * dx + dy * dy + dz * dz;
    if (d <= r2) {
      if (!got) { first = j; got = true; }
      nbr[(size_t)t * k + cnt] = j; cnt++;
    }
  }
  for (; cnt < k; cnt++) nbr[(size_t)t * k + cnt] = first;
}

// -------------------- build grouped f16 tensor G[M, Kp]: [feat | rel_xyz | zero pad] ----------
__global__ void k_group(const float* __restrict__ pts, const float* __restrict__ xyz,
                        const float* __restrict__ nx, const int* __restrict__ nbr,
                        int S, int k, int Cf, int Nin, int Kp, _Float16* __restrict__ G) {
  size_t i = (size_t)blockIdx.x * blockDim.x + threadIdx.x;
  size_t total = (size_t)BB * S * k * Kp;
  if (i >= total) return;
  int c = (int)(i % Kp);
  size_t row = i / Kp;
  int j  = (int)(row % k);
  int bs = (int)(row / k);
  int b = bs / S;
  int id = nbr[(size_t)bs * k + j];
  float v = 0.0f;
  int cin = Cf + 3;
  if (c < Cf)       v = pts[((size_t)b * Nin + id) * Cf + c];
  else if (c < cin) v = xyz[((size_t)b * Nin + id) * 3 + (c - Cf)] - nx[((size_t)bs) * 3 + (c - Cf)];
  G[i] = (_Float16)v;
}

// -------------------- pad/convert weights: W (Cout,Cin) f32 -> Wh (Np,Kp) f16 --------------------
__global__ void k_prepw(const float* __restrict__ W, int Cout, int Cin, int Np, int Kp,
                        _Float16* __restrict__ Wh) {
  int i = blockIdx.x * blockDim.x + threadIdx.x;
  if (i >= Np * Kp) return;
  int c = i % Kp, n = i / Kp;
  float v = (n < Cout && c < Cin) ? W[(size_t)n * Cin + c] : 0.0f;
  Wh[i] = (_Float16)v;
}

// -------------------- LDS-staged WMMA GEMM --------------------
// Block tile 128M x 64N, 8 waves, each wave: one A fragment reused across 4 accumulators.
// Y[M,Cout] = G[M,Kp] x Wh[Np64,Kp]^T + bias   (Kp % 32 == 0, M % 128 == 0, Np64 % 64 == 0)
__global__ __launch_bounds__(256) void k_wmma_gemm(const _Float16* __restrict__ G,
                                                   const _Float16* __restrict__ Wh,
                                                   const float* __restrict__ bias,
                                                   float* __restrict__ Y,
                                                   int M, int Kp, int Cout) {
  __shared__ _Float16 sA[128 * 32];   // 128 rows x 32 K halves  (8 KB)
  __shared__ _Float16 sB[64 * 32];    // 64 cols  x 32 K halves  (4 KB)
  const int tid  = threadIdx.x;
  const int wave = tid >> 5;
  const int lane = tid & 31;
  const int m0 = blockIdx.x * 128;
  const int n0 = blockIdx.y * 64;
  const v8f z = {0.f, 0.f, 0.f, 0.f, 0.f, 0.f, 0.f, 0.f};
  v8f acc[4] = {z, z, z, z};

  for (int kb = 0; kb < Kp; kb += 32) {
#ifdef USE_ASYNC_LDS
    // Per-lane B128 async copies straight into LDS (ASYNCcnt-tracked).
    for (int i = tid; i < 512; i += 256) {        // A: 128 rows x 4 16B segments
      int r = i >> 2, seg = i & 3;
      const _Float16* gp = G + (size_t)(m0 + r) * Kp + kb + seg * 8;
      __builtin_amdgcn_global_load_async_to_lds_b128(
          (__attribute__((address_space(1))) v4i32*)gp,
          (__attribute__((address_space(3))) v4i32*)(sA + r * 32 + seg * 8), 0, 0);
    }
    {                                             // B: 64 cols x 4 16B segments
      int n = tid >> 2, seg = tid & 3;
      const _Float16* wp = Wh + (size_t)(n0 + n) * Kp + kb + seg * 8;
      __builtin_amdgcn_global_load_async_to_lds_b128(
          (__attribute__((address_space(1))) v4i32*)wp,
          (__attribute__((address_space(3))) v4i32*)(sB + n * 32 + seg * 8), 0, 0);
    }
    __builtin_amdgcn_s_wait_asynccnt(0);
#else
    for (int i = tid; i < 128 * 16; i += 256) {   // A tile (uint = 2 halves)
      int r = i >> 4, c2 = i & 15;
      unsigned int u = *(const unsigned int*)(G + (size_t)(m0 + r) * Kp + kb + c2 * 2);
      *(unsigned int*)(sA + r * 32 + c2 * 2) = u;
    }
    for (int i = tid; i < 64 * 16; i += 256) {    // B tile
      int n = i >> 4, c2 = i & 15;
      unsigned int u = *(const unsigned int*)(Wh + (size_t)(n0 + n) * Kp + kb + c2 * 2);
      *(unsigned int*)(sB + n * 32 + c2 * 2) = u;
    }
#endif
    if (kb + 32 < Kp)  // hint next A tile toward the caches
      __builtin_prefetch(G + (size_t)(m0 + (tid >> 1)) * Kp + kb + 32, 0, 1);
    __syncthreads();

    // A fragment: lane -> row (lane&15), interleaved K map, upper half-wave = K+8
    v16h a;
    const int ar = wave * 16 + (lane & 15);
    const int ka = (lane >> 4) << 3;
#pragma unroll
    for (int v = 0; v < 8; v++) {
      int kk = ((v & 3) * 2) + ((v >> 2) * 16) + ka;
      a[2 * v]     = sA[ar * 32 + kk];
      a[2 * v + 1] = sA[ar * 32 + kk + 1];
    }
    // B fragments: lane -> col (lane&15), interleaved K map matching A, upper half-wave = K+8
    const int bc = (lane & 15);
    const int kb16 = (lane >> 4) << 3;
#pragma unroll
    for (int t = 0; t < 4; t++) {
      v16h bfr;
#pragma unroll
      for (int h = 0; h < 16; h++) bfr[h] = sB[(t * 16 + bc) * 32 + kb16 + (h & 7) + ((h >> 3) << 4)];
      acc[t] = __builtin_amdgcn_wmma_f32_16x16x32_f16(false, a, false, bfr,
                                                      (short)0, acc[t], false, false);
    }
    __syncthreads();
  }

  const int mbase = m0 + wave * 16 + ((lane >> 4) << 3);
#pragma unroll
  for (int t = 0; t < 4; t++) {
    const int oc = n0 + t * 16 + (lane & 15);
    if (oc < Cout) {
      const float bv = bias[oc];
#pragma unroll
      for (int r = 0; r < 8; r++) {
        int m = mbase + r;
        if (m < M) Y[(size_t)m * Cout + oc] = acc[t][r] + bv;
      }
    }
  }
}

// -------------------- per-channel batch stats over all M rows --------------------
__global__ __launch_bounds__(256) void k_stats(const float* __restrict__ Y, int M, int C,
                                               float* __restrict__ mean, float* __restrict__ rstd) {
  __shared__ float ss[256], sq[256];
  int o = blockIdx.x, tid = threadIdx.x;
  float s = 0.f, q = 0.f;
  for (int m = tid; m < M; m += 256) { float v = Y[(size_t)m * C + o]; s += v; q += v * v; }
  ss[tid] = s; sq[tid] = q;
  __syncthreads();
  for (int st = 128; st > 0; st >>= 1) {
    if (tid < st) { ss[tid] += ss[tid + st]; sq[tid] += sq[tid + st]; }
    __syncthreads();
  }
  if (tid == 0) {
    float mu = ss[0] / (float)M;
    float var = sq[0] / (float)M - mu * mu;
    mean[o] = mu;
    rstd[o] = rsqrtf(var + 1e-5f);
  }
}

// -------------------- BN + ReLU in place; optionally emit next layer's f16 K-padded input ------
__global__ void k_bnrelu(float* __restrict__ Y, int M, int Cout, int Kp,
                         const float* __restrict__ mean, const float* __restrict__ rstd,
                         const float* __restrict__ gamma, const float* __restrict__ beta,
                         _Float16* __restrict__ Gn) {
  size_t i = (size_t)blockIdx.x * blockDim.x + threadIdx.x;
  if (i >= (size_t)M * Kp) return;
  int o = (int)(i % Kp);
  size_t m = i / Kp;
  if (o < Cout) {
    float v = Y[m * Cout + o];
    v = (v - mean[o]) * rstd[o] * gamma[o] + beta[o];
    v = fmaxf(v, 0.0f);
    Y[m * Cout + o] = v;
    if (Gn) Gn[m * Kp + o] = (_Float16)v;
  } else if (Gn) {
    Gn[m * Kp + o] = (_Float16)0.0f;
  }
}

// -------------------- max over k neighbors, write into concat output --------------------
__global__ void k_maxpool(const float* __restrict__ Y, int S, int k, int C, int Ctot, int coff,
                          float* __restrict__ out) {
  int t = blockIdx.x * blockDim.x + threadIdx.x;
  if (t >= BB * S * C) return;
  int o = t % C;
  int bs = t / C;
  float m = -3.4e38f;
  const float* yp = Y + ((size_t)bs * k) * C + o;
  for (int j = 0; j < k; j++) m = fmaxf(m, yp[(size_t)j * C]);
  out[(size_t)bs * Ctot + coff + o] = m;
}

// ==================== host orchestration ====================
extern "C" void kernel_launch(void* const* d_in, const int* in_sizes, int n_in,
                              void* d_out, int out_size, void* d_ws, size_t ws_size,
                              hipStream_t stream) {
  (void)in_sizes; (void)n_in; (void)out_size; (void)ws_size;
  const float* pc = (const float*)d_in[0];
  // pytree order: pc, then per layer {W, b, beta, gamma} (alphabetical), layers in
  // stage-major / branch / layer order.
  auto P = [&](int st, int br, int ly, int q) -> const float* {
    return (const float*)d_in[1 + (((st * 2 + br) * 3) + ly) * 4 + q];
  };
  static const int   npts[4]     = {2048, 512, 128, 64};
  static const float radii[4][2] = {{0.05f, 0.1f}, {0.1f, 0.2f}, {0.2f, 0.4f}, {0.4f, 0.8f}};
  static const int   kk[2]       = {16, 32};
  static const int   mlps[4][2][3] = {
      {{16, 16, 32},   {32, 32, 64}},
      {{64, 64, 128},  {64, 96, 128}},
      {{128, 196, 256},{128, 196, 256}},
      {{256, 256, 256},{256, 256, 256}}};
  static const int ctot[4] = {96, 256, 512, 512};

  char* w = (char*)d_ws;
  size_t off = 0;
  auto carve = [&](size_t bytes) -> char* {
    char* p = w + off;
    off += (bytes + 255) & ~(size_t)255;
    return p;
  };
  float*    xyz0  = (float*)carve((size_t)BB * NN * 3 * 4);
  float*    nxA   = (float*)carve((size_t)BB * 2048 * 3 * 4);
  float*    nxB   = (float*)carve((size_t)BB * 2048 * 3 * 4);
  int*      fpsi  = (int*)carve((size_t)BB * 2048 * 4);
  int*      nbr   = (int*)carve((size_t)BB * 2048 * 32 * 4);
  _Float16* Wh    = (_Float16*)carve((size_t)256 * 544 * 2);
  float*    meanb = (float*)carve(1024);
  float*    rstdb = (float*)carve(1024);
  _Float16* Ga    = (_Float16*)carve((size_t)16 * 1024 * 1024);
  _Float16* Gb    = (_Float16*)carve((size_t)16 * 1024 * 1024);
  float*    Ybuf  = (float*)carve((size_t)64 * 1024 * 1024);

  k_split_pc<<<(BB * NN + 255) / 256, 256, 0, stream>>>(pc, xyz0);

  const float* xyz_cur = xyz0; int Nin = NN;
  const float* pts_cur = xyz0; int Cf = 3;
  float* outp = (float*)d_out;
  size_t ooff = 0;

  for (int st = 0; st < 4; st++) {
    int S = npts[st];
    float* nx = (st & 1) ? nxB : nxA;
    k_fps<<<BB, 256, 0, stream>>>(xyz_cur, Nin, S, fpsi);
    k_gather_nx<<<(BB * S + 255) / 256, 256, 0, stream>>>(xyz_cur, fpsi, Nin, S, nx);
    float* stage_out = outp + ooff;
    int coff = 0;
    for (int br = 0; br < 2; br++) {
      int kn = kk[br];
      float r = radii[st][br];
      int M = BB * S * kn;
      k_ballq<<<(BB * S + 255) / 256, 256, 0, stream>>>(xyz_cur, nx, Nin, S, r * r, kn, nbr);
      int cin = Cf + 3;
      int Kp = alignup(cin, 32);
      k_group<<<(unsigned)(((size_t)M * Kp + 255) / 256), 256, 0, stream>>>(
          pts_cur, xyz_cur, nx, nbr, S, kn, Cf, Nin, Kp, Ga);
      _Float16* Gin = Ga;
      _Float16* Gnx = Gb;
      int cprev = cin;
      for (int ly = 0; ly < 3; ly++) {
        int cout = mlps[st][br][ly];
        int Kp2  = alignup(cprev, 32);
        int Np64 = alignup(cout, 64);
        k_prepw<<<(Np64 * Kp2 + 255) / 256, 256, 0, stream>>>(P(st, br, ly, 0), cout, cprev, Np64, Kp2, Wh);
        dim3 gg((unsigned)(M / 128), (unsigned)(Np64 / 64));
        k_wmma_gemm<<<gg, 256, 0, stream>>>(Gin, Wh, P(st, br, ly, 1), Ybuf, M, Kp2, cout);
        k_stats<<<cout, 256, 0, stream>>>(Ybuf, M, cout, meanb, rstdb);
        bool last = (ly == 2);
        int KpO = last ? cout : alignup(cout, 32);
        k_bnrelu<<<(unsigned)(((size_t)M * KpO + 255) / 256), 256, 0, stream>>>(
            Ybuf, M, cout, KpO, meanb, rstdb, P(st, br, ly, 3), P(st, br, ly, 2),
            last ? (_Float16*)nullptr : Gnx);
        if (!last) { _Float16* t = Gin; Gin = Gnx; Gnx = t; }
        cprev = cout;
      }
      k_maxpool<<<(unsigned)(((size_t)BB * S * cprev + 255) / 256), 256, 0, stream>>>(
          Ybuf, S, kn, cprev, ctot[st], coff, stage_out);
      coff += cprev;
    }
    ooff += (size_t)BB * S * ctot[st];
    pts_cur = stage_out;
    Cf = ctot[st];
    xyz_cur = nx;
    Nin = S;
  }
}
